// HierarchicalContrastiveLossAug_24335284699668
// MI455X (gfx1250) — matrix operand order repaired and checked by reading, not compiled, not run on hardware
//
#include <hip/hip_runtime.h>
#include <hip/hip_bf16.h>
#include <math.h>

typedef __attribute__((ext_vector_type(16))) _Float16 v16h;
typedef __attribute__((ext_vector_type(8)))  _Float16 v8h;
typedef __attribute__((ext_vector_type(4)))  _Float16 v4h;
typedef __attribute__((ext_vector_type(8)))  float    v8f;

#define ALPHA_W 0.5f
#define NEG_BIG (-1e30f)

// ---------------- wave32 reductions ----------------
__device__ __forceinline__ float wave_max32(float v) {
#pragma unroll
  for (int off = 16; off > 0; off >>= 1) v = fmaxf(v, __shfl_xor(v, off, 32));
  return v;
}
__device__ __forceinline__ float wave_sum32(float v) {
#pragma unroll
  for (int off = 16; off > 0; off >>= 1) v += __shfl_xor(v, off, 32);
  return v;
}

// ---------------- tiny utility kernels ----------------
__global__ void k_zero(float* acc) { *acc = 0.f; }
__global__ void k_write(const float* __restrict__ acc, float* __restrict__ out) { out[0] = acc[0]; }

// Global min/max of t (single block). diff.max() == max(t) - min(t).
__global__ void k_minmax(const float* __restrict__ t, int N, float* __restrict__ mm) {
  __shared__ float smin[256], smax[256];
  float lmin = INFINITY, lmax = -INFINITY;
  for (int i = threadIdx.x; i < N; i += 256) {
    float v = t[i];
    lmin = fminf(lmin, v);
    lmax = fmaxf(lmax, v);
  }
  smin[threadIdx.x] = lmin; smax[threadIdx.x] = lmax;
  __syncthreads();
  for (int s = 128; s > 0; s >>= 1) {
    if ((int)threadIdx.x < s) {
      smin[threadIdx.x] = fminf(smin[threadIdx.x], smin[threadIdx.x + s]);
      smax[threadIdx.x] = fmaxf(smax[threadIdx.x], smax[threadIdx.x + s]);
    }
    __syncthreads();
  }
  if (threadIdx.x == 0) { mm[0] = smin[0]; mm[1] = smax[0]; }
}

// ---------------- instance loss: per-row top-16 nearest timestamps ----------------
__global__ void k_topk(const float* __restrict__ t, const float* __restrict__ mm,
                       int N, int lg, int* __restrict__ idx) {
  int n = blockIdx.x * blockDim.x + threadIdx.x;
  if (n >= N) return;
  float tn  = t[n];
  float pen = (mm[1] - mm[0]) + 0.001f;   // diff.max() + 0.001
  int seq = n >> lg;                      // sequence id (T = 1<<lg)
  float bd[16]; int bi[16];
#pragma unroll
  for (int j = 0; j < 16; ++j) { bd[j] = INFINITY; bi[j] = 0; }
  for (int m = 0; m < N; ++m) {
    float dd = fabsf(tn - t[m]);
    if ((m >> lg) == seq) dd += pen;      // same-sequence penalty
    if (dd < bd[15]) {
      float cd = dd; int ci = m;
#pragma unroll
      for (int j = 0; j < 16; ++j) {
        if (cd < bd[j]) {
          float td = bd[j]; int ti = bi[j];
          bd[j] = cd; bi[j] = ci; cd = td; ci = ti;
        }
      }
    }
  }
#pragma unroll
  for (int j = 0; j < 16; ++j) idx[(size_t)n * 16 + j] = bi[j];
}

// One wave per row: lane j<16 -> dot(z1[n], z1[idx[j]]); lane j>=16 -> dot(z1[n], z2[idx[j-16]]).
__global__ void k_instance(const float* __restrict__ z1f, const float* __restrict__ z2f,
                           const int* __restrict__ idx, float scale, float* __restrict__ acc) {
  int n = blockIdx.x;
  int lane = threadIdx.x;
  int nb = idx[(size_t)n * 16 + (lane & 15)];
  const float* zsel = (lane < 16) ? z1f : z2f;
  const float4* a  = (const float4*)(z1f + (size_t)n * 64);
  const float4* bv = (const float4*)(zsel + (size_t)nb * 64);
  const float4* p2 = (const float4*)(z2f + (size_t)n * 64);
  float dot = 0.f, pos = 0.f;
#pragma unroll
  for (int k = 0; k < 16; ++k) {
    float4 av = a[k], bb = bv[k], pv = p2[k];
    dot += av.x * bb.x + av.y * bb.y + av.z * bb.z + av.w * bb.w;
    pos += av.x * pv.x + av.y * pv.y + av.z * pv.z + av.w * pv.w;  // uniform across lanes
  }
  float m = fmaxf(pos, wave_max32(dot));
  float s = wave_sum32(__expf(dot - m)) + __expf(pos - m);
  if (lane == 0) atomicAdd(acc, -scale * (pos - m - __logf(s)));
}

// ---------------- f32 -> packed f16 Z = [z1;z2] mirror for the WMMA kernel ----------------
__global__ void k_cvt(const float* __restrict__ z1, const float* __restrict__ z2,
                      _Float16* __restrict__ Zh, int T, int twoT, int total4 /* B*2T*16 */) {
  int e = blockIdx.x * blockDim.x + threadIdx.x;
  if (e >= total4) return;
  int k4 = e & 15;
  int rr = e >> 4;
  int r  = rr % twoT;
  int b  = rr / twoT;
  const float* src = (r < T) ? (z1 + ((size_t)b * T + r) * 64)
                             : (z2 + ((size_t)b * T + (r - T)) * 64);
  float4 f = *(const float4*)(src + k4 * 4);
  v4h h;
  h[0] = (_Float16)f.x; h[1] = (_Float16)f.y; h[2] = (_Float16)f.z; h[3] = (_Float16)f.w;
  *(v4h*)(Zh + ((size_t)b * twoT + r) * 64 + k4 * 4) = h;
}

// ---------------- temporal loss: fused Z Z^T (WMMA f16) + branchless online logsumexp ----
// One wave owns a 32-row tile (two 16-row A fragments); each B fragment feeds 4 WMMAs.
// Row-validity handled only at write-out; hot loop is exec-mask-free (cndmask only).
__global__ void __launch_bounds__(32)
k_temporal(const _Float16* __restrict__ Zh, int T, int twoT, float scale,
           float* __restrict__ acc) {
  const int lane = threadIdx.x;
  const int b  = blockIdx.y;
  const int rt = blockIdx.x;          // 32-row tile index
  const int hrow = lane & 15;         // row-in-tile (A) / col-in-tile (B,C,D)
  const int hgrp = lane >> 4;         // K-half group
  const _Float16* Zb = Zh + (size_t)b * twoT * 64;

  // Two A fragments (rows rt*32 + s*16 + hrow), packed per ISA 16-bit A layout.
  v16h a0[2], a1[2];
#pragma unroll
  for (int s = 0; s < 2; ++s) {
    int r  = rt * 32 + s * 16 + hrow;
    int rc = r < twoT ? r : twoT - 1;
    const _Float16* arow = Zb + (size_t)rc * 64;
    v8h c0 = *(const v8h*)(arow + hgrp * 8);
    v8h c1 = *(const v8h*)(arow + 16 + hgrp * 8);
    v8h c2 = *(const v8h*)(arow + 32 + hgrp * 8);
    v8h c3 = *(const v8h*)(arow + 48 + hgrp * 8);
    a0[s] = __builtin_shufflevector(c0, c1, 0,1,2,3,4,5,6,7,8,9,10,11,12,13,14,15);
    a1[s] = __builtin_shufflevector(c2, c3, 0,1,2,3,4,5,6,7,8,9,10,11,12,13,14,15);
  }

  const int mbase = hgrp * 8;
  float rm[16], rs[16], rp[16];
  int rowg[16], partner[16];
#pragma unroll
  for (int i = 0; i < 16; ++i) {
    int s = i >> 3, v = i & 7;
    rowg[i]    = rt * 32 + s * 16 + mbase + v;
    partner[i] = (rowg[i] < T) ? (rowg[i] + T) : (rowg[i] - T);
    rm[i] = NEG_BIG; rs[i] = 0.f; rp[i] = NEG_BIG;
  }

  const int ntiles = (twoT + 15) >> 4;
  for (int jt = 0; jt < ntiles; ++jt) {
    int colg = jt * 16 + hrow;
    bool cval = colg < twoT;
    int cc = cval ? colg : twoT - 1;
    const _Float16* crow = Zb + (size_t)cc * 64;
    v8h d0 = *(const v8h*)(crow + hgrp * 8);
    v8h d1 = *(const v8h*)(crow + 16 + hgrp * 8);
    v8h d2 = *(const v8h*)(crow + 32 + hgrp * 8);
    v8h d3 = *(const v8h*)(crow + 48 + hgrp * 8);
    v16h b0 = __builtin_shufflevector(d0, d1, 0,1,2,3,4,5,6,7,8,9,10,11,12,13,14,15);        // ISA 16-bit B layout
    v16h b1 = __builtin_shufflevector(d2, d3, 0,1,2,3,4,5,6,7,8,9,10,11,12,13,14,15);

#pragma unroll
    for (int s = 0; s < 2; ++s) {
      v8f c = {};
      c = __builtin_amdgcn_wmma_f32_16x16x32_f16(false, a0[s], false, b0, (short)0, c, false, false);
      c = __builtin_amdgcn_wmma_f32_16x16x32_f16(false, a1[s], false, b1, (short)0, c, false, false);
#pragma unroll
      for (int v = 0; v < 8; ++v) {
        int i = s * 8 + v;
        float x  = c[v];
        bool ok  = cval && (colg != rowg[i]);          // diagonal excluded
        float xm = ok ? x : NEG_BIG;
        rp[i] = (colg == partner[i]) ? x : rp[i];      // positive logit
        float nm = fmaxf(rm[i], xm);                   // branchless online logsumexp
        rs[i] = rs[i] * __expf(rm[i] - nm) + __expf(xm - nm);
        rm[i] = nm;
      }
    }
  }

  // merge (m, s, pos) across the 16 lanes of each half-group
#pragma unroll
  for (int i = 0; i < 16; ++i) {
    float m = rm[i], sv = rs[i], p = rp[i];
#pragma unroll
    for (int off = 1; off < 16; off <<= 1) {
      float mo = __shfl_xor(m, off, 32);
      float so = __shfl_xor(sv, off, 32);
      float po = __shfl_xor(p, off, 32);
      float nm = fmaxf(m, mo);
      sv = sv * __expf(m - nm) + so * __expf(mo - nm);
      m  = nm;
      p  = fmaxf(p, po);
    }
    rm[i] = m; rs[i] = sv; rp[i] = p;
  }

  if (hrow == 0) {
    float local = 0.f;
#pragma unroll
    for (int i = 0; i < 16; ++i)
      if (rowg[i] < twoT) local += rp[i] - rm[i] - __logf(rs[i]);
    atomicAdd(acc, -scale * local);
  }
}

// ---------------- hierarchy pooling ----------------
__global__ void k_pool_z(const float* __restrict__ in, float* __restrict__ out,
                         int Tin, int total /* B*Tout*64 */) {
  int e = blockIdx.x * blockDim.x + threadIdx.x;
  if (e >= total) return;
  int Tout = Tin >> 1;
  int c = e & 63;
  int r = e >> 6;
  int i = r % Tout;
  int b = r / Tout;
  size_t base = ((size_t)b * Tin + 2 * i) * 64 + c;
  out[e] = fmaxf(in[base], in[base + 64]);
}
__global__ void k_pool_t(const float* __restrict__ in, float* __restrict__ out,
                         int Tin, int total /* B*Tout */) {
  int e = blockIdx.x * blockDim.x + threadIdx.x;
  if (e >= total) return;
  int Tout = Tin >> 1;
  int i = e % Tout;
  int b = e / Tout;
  out[e] = 0.5f * (in[(size_t)b * Tin + 2 * i] + in[(size_t)b * Tin + 2 * i + 1]);
}

// ---------------- host orchestration ----------------
extern "C" void kernel_launch(void* const* d_in, const int* in_sizes, int n_in,
                              void* d_out, int out_size, void* d_ws, size_t ws_size,
                              hipStream_t stream) {
  (void)n_in; (void)out_size; (void)ws_size;
  const float* z1_in = (const float*)d_in[0];
  const float* z2_in = (const float*)d_in[1];
  const float* t_in  = (const float*)d_in[2];

  const int B  = 16;
  const int T0 = in_sizes[2] / B;   // 1024

  int d = 1;
  for (int tt = T0; tt > 1; tt >>= 1) ++d;     // 11 levels
  const float invd = 1.0f / (float)d;

  // workspace layout (float units, 1KB-aligned blocks)
  float* ws   = (float*)d_ws;
  float* acc  = ws;                                   // 1 f32 accumulator
  float* mm   = ws + 64;                              // [min, max] of t (per level, overwritten)
  int*   idxb = (int*)(ws + 256);                     // B*T0*16 ints (reused per level)
  size_t idxcap = (size_t)B * T0 * 16;                // 262144
  size_t zcap   = (size_t)B * (T0 / 2) * 64;          // 524288 (ping-pong)
  float* z1A = (float*)(idxb + idxcap);
  float* z1B = z1A + zcap;
  float* z2A = z1B + zcap;
  float* z2B = z2A + zcap;
  float* tA  = z2B + zcap;
  float* tB  = tA + (size_t)B * (T0 / 2);
  _Float16* Zh = (_Float16*)(tB + (size_t)B * (T0 / 2));  // B*2*T0*64 halfs (reused per level)

  k_zero<<<1, 1, 0, stream>>>(acc);

  const float* cz1 = z1_in;
  const float* cz2 = z2_in;
  const float* ct  = t_in;
  int T = T0, pp = 0;

  for (;;) {
    int N = B * T;
    int lg = 0; while ((1 << lg) < T) ++lg;

    // instance contrastive loss (alpha term)
    k_minmax<<<1, 256, 0, stream>>>(ct, N, mm);
    k_topk<<<(N + 127) / 128, 128, 0, stream>>>(ct, mm, N, lg, idxb);
    k_instance<<<N, 32, 0, stream>>>(cz1, cz2, idxb, ALPHA_W * invd / (float)N, acc);

    if (T == 1) break;

    // temporal contrastive loss ((1-alpha) term): f16 mirror + fused WMMA logsumexp
    int twoT = 2 * T;
    int total4 = B * twoT * 16;
    k_cvt<<<(total4 + 255) / 256, 256, 0, stream>>>(cz1, cz2, Zh, T, twoT, total4);
    dim3 tg((twoT + 31) / 32, B);
    k_temporal<<<tg, 32, 0, stream>>>(Zh, T, twoT,
                                      (1.0f - ALPHA_W) * invd / (float)(2 * B * T), acc);

    // pool to next level
    int Tn = T >> 1;
    int ztotal = B * Tn * 64, ttotal = B * Tn;
    float* nz1 = (pp == 0) ? z1A : z1B;
    float* nz2 = (pp == 0) ? z2A : z2B;
    float* nt  = (pp == 0) ? tA  : tB;
    k_pool_z<<<(ztotal + 255) / 256, 256, 0, stream>>>(cz1, nz1, T, ztotal);
    k_pool_z<<<(ztotal + 255) / 256, 256, 0, stream>>>(cz2, nz2, T, ztotal);
    k_pool_t<<<(ttotal + 255) / 256, 256, 0, stream>>>(ct, nt, T, ttotal);
    cz1 = nz1; cz2 = nz2; ct = nt;
    T = Tn; pp ^= 1;
  }

  k_write<<<1, 1, 0, stream>>>(acc, (float*)d_out);
}
